// dis_VRNN_82952998355887
// MI455X (gfx1250) — matrix operand order repaired and checked by reading, not compiled, not run on hardware
//
#include <hip/hip_runtime.h>

// VRNN forward for MI455X (gfx1250).
// f16 WMMA GEMM (128x64 block tile, 2x2 wmma/wave, double-buffered LDS,
// async global->LDS when available) + fused concat/head layout.
// T=64, B=512, X=784(->800/832), F=Z=64, H=512, 3H=1536.

typedef _Float16 f16;
typedef __attribute__((ext_vector_type(16))) _Float16 v16h;
typedef __attribute__((ext_vector_type(8)))  _Float16 v8h;
typedef __attribute__((ext_vector_type(8)))  float    v8f;
typedef __attribute__((ext_vector_type(4)))  int      v4i;

#define T_DIM  64
#define B_DIM  512
#define X_DIM  784
#define XP_DIM 800
#define XO_DIM 832   /* output-padded X (mult of 64) */
#define F_DIM  64
#define Z_DIM  64
#define H_DIM  512
#define H3_DIM 1536
#define TB_DIM (T_DIM * B_DIM)
#define UC_LD  2560  /* ucat row: [px | phi_z | phi_f | h | px2] */

#if defined(__HIP_DEVICE_COMPILE__) && __has_builtin(__builtin_amdgcn_global_load_async_to_lds_b128)
#define ASYNC_LDS 1
#else
#define ASYNC_LDS 0
#endif

typedef __attribute__((address_space(1))) v4i as1_v4i;  // global 128-bit chunk
typedef __attribute__((address_space(3))) v4i as3_v4i;  // LDS 128-bit chunk

// 16-byte global -> LDS copy (one b128 lane-worth per call)
__device__ __forceinline__ void lds_cp16(const f16* g, f16* l) {
#if ASYNC_LDS
  __builtin_amdgcn_global_load_async_to_lds_b128(
      (as1_v4i*)(size_t)g, (as3_v4i*)(unsigned)(size_t)l, 0, 0);
#else
  *(v8h*)l = *(const v8h*)g;
#endif
}
__device__ __forceinline__ void stage_fence() {
#if ASYNC_LDS
#if __has_builtin(__builtin_amdgcn_s_wait_asynccnt)
  __builtin_amdgcn_s_wait_asynccnt(0);
#else
  asm volatile("s_wait_asynccnt 0x0" ::: "memory");
#endif
#endif
}

// A fragment (16x32 f16, ISA 7.12.2): elems 0..7 -> K=hid*8.., 8..15 -> K=16+hid*8..
__device__ __forceinline__ v16h frag_a(const f16* row, int hid) {
  const v8h lo = *(const v8h*)(row + (hid << 3));
  const v8h hi = *(const v8h*)(row + 16 + (hid << 3));
  v16h r;
#pragma unroll
  for (int j = 0; j < 8; ++j) { r[j] = lo[j]; r[j + 8] = hi[j]; }
  return r;
}
// B fragment (32x16 f16): col = lane&15, elems 0..7 -> K=hid*8.., 8..15 -> K=16+hid*8.. (row of W[N,K])
__device__ __forceinline__ v16h frag_b(const f16* row, int hid) {
  const v8h lo = *(const v8h*)(row + (hid << 3));
  const v8h hi = *(const v8h*)(row + 16 + (hid << 3));
  v16h r;
#pragma unroll
  for (int j = 0; j < 8; ++j) { r[j] = lo[j]; r[j + 8] = hi[j]; }
  return r;
}
__device__ __forceinline__ v8f wmma16(const v16h a, const v16h b, const v8f c) {
  return __builtin_amdgcn_wmma_f32_16x16x32_f16(false, a, false, b, (short)0, c,
                                                false, false);
}

// ------------------------------------------------------------------
// GEMM: C[M,N] = act( A[M,K](f16) * W[N,K](f16)^T + bias )
// Block 256 thr = 8 waves; tile 128x64; wave tile 32x32 (2x2 wmma).
// M%128==0, N%64==0, K%32==0, lda%8==0, ldw%8==0.
// ------------------------------------------------------------------
__launch_bounds__(256)
__global__ void wmma_gemm_kernel(const f16* __restrict__ A, int lda,
                                 const f16* __restrict__ W, int ldw,
                                 const float* __restrict__ bias,
                                 float* __restrict__ C32,
                                 f16* __restrict__ C16,
                                 int ldc, int K, int act) {
  __shared__ f16 As[2][128][32];
  __shared__ f16 Ws[2][64][32];

  const int tid  = threadIdx.x;
  const int lane = tid & 31;
  const int wv   = tid >> 5;
  const int m_off = (wv & 3) << 5;   // 0,32,64,96
  const int n_off = (wv >> 2) << 5;  // 0,32
  const int m0 = blockIdx.y << 7;
  const int n0 = blockIdx.x << 6;
  const int hid = lane >> 4;
  const int l15 = lane & 15;

  // staging: A tile 128x32 -> 2 chunks/thread; W tile 64x32 -> 1 chunk/thread
  const int ar0 = (2 * tid) >> 2,     ac0 = ((2 * tid) & 3) << 3;
  const int ar1 = (2 * tid + 1) >> 2, ac1 = ((2 * tid + 1) & 3) << 3;
  const int wr  = tid >> 2,           wc  = (tid & 3) << 3;
  const f16* gA0 = A + (size_t)(m0 + ar0) * lda + ac0;
  const f16* gA1 = A + (size_t)(m0 + ar1) * lda + ac1;
  const f16* gW  = W + (size_t)(n0 + wr) * ldw + wc;

  v8f acc00 = {}, acc01 = {}, acc10 = {}, acc11 = {};
  if (bias) {
    const float b0 = bias[n0 + n_off + l15];
    const float b1 = bias[n0 + n_off + 16 + l15];
#pragma unroll
    for (int r = 0; r < 8; ++r) {
      acc00[r] = b0; acc10[r] = b0; acc01[r] = b1; acc11[r] = b1;
    }
  }

  auto stage = [&](int buf, int k0) {
    lds_cp16(gA0 + k0, &As[buf][ar0][ac0]);
    lds_cp16(gA1 + k0, &As[buf][ar1][ac1]);
    lds_cp16(gW + k0, &Ws[buf][wr][wc]);
  };

  stage(0, 0);
  stage_fence();
  __syncthreads();

  int buf = 0;
  for (int k0 = 0; k0 < K; k0 += 32) {
    if (k0 + 32 < K) stage(buf ^ 1, k0 + 32);  // DMA next tile during compute

    const v16h a0 = frag_a(&As[buf][m_off + l15][0], hid);
    const v16h a1 = frag_a(&As[buf][m_off + 16 + l15][0], hid);
    const v16h b0 = frag_b(&Ws[buf][n_off + l15][0], hid);
    const v16h b1 = frag_b(&Ws[buf][n_off + 16 + l15][0], hid);
    acc00 = wmma16(a0, b0, acc00);
    acc01 = wmma16(a0, b1, acc01);
    acc10 = wmma16(a1, b0, acc10);
    acc11 = wmma16(a1, b1, acc11);

    stage_fence();
    __syncthreads();
    buf ^= 1;
  }

  auto emit = [&](const v8f& a, int gm, int gc) {
#pragma unroll
    for (int r = 0; r < 8; ++r) {
      float v = a[r];
      if (act == 1) v = fmaxf(v, 0.0f);
      const size_t o = (size_t)(gm + r) * ldc + gc;
      if (C32) C32[o] = v;
      if (C16) C16[o] = (f16)v;
    }
  };
  const int gm0 = m0 + m_off + (hid << 3);
  const int gc0 = n0 + n_off + l15;
  emit(acc00, gm0, gc0);
  emit(acc01, gm0, gc0 + 16);
  emit(acc10, gm0 + 16, gc0);
  emit(acc11, gm0 + 16, gc0 + 16);
}

// ------------------------------------------------------------------
// Elementwise / reduction kernels
// ------------------------------------------------------------------
__device__ __forceinline__ float softplusf(float x) {
  return (x > 20.0f) ? x : log1pf(expf(x));
}
__device__ __forceinline__ float sigmoidf(float x) {
  return 1.0f / (1.0f + expf(-x));
}
__device__ __forceinline__ void block_reduce_add(float v, float* target) {
  __shared__ float red[256];
  red[threadIdx.x] = v;
  __syncthreads();
  for (int s = 128; s > 0; s >>= 1) {
    if ((int)threadIdx.x < s) red[threadIdx.x] += red[threadIdx.x + s];
    __syncthreads();
  }
  if (threadIdx.x == 0) atomicAdd(target, red[0]);
}

__global__ void zero_f32_kernel(float* p, long long n) {
  long long i = (long long)blockIdx.x * blockDim.x + threadIdx.x;
  if (i < n) p[i] = 0.0f;
}
__global__ void zero_f16_kernel(f16* p, long long n) {
  long long i = (long long)blockIdx.x * blockDim.x + threadIdx.x;
  if (i < n) p[i] = (f16)0.0f;
}
// f32 [rows,cols] (ld=sld) -> f16 rectangle [dRows,dCols] at dst (ld=dld), zero-padded
__global__ void pack_w_kernel(const float* __restrict__ src, int sld, int rows, int cols,
                              f16* __restrict__ dst, int dld, int dCols, long long total) {
  long long i = (long long)blockIdx.x * blockDim.x + threadIdx.x;
  if (i >= total) return;
  int r = (int)(i / dCols);
  int c = (int)(i % dCols);
  float v = (r < rows && c < cols) ? src[(size_t)r * sld + c] : 0.0f;
  dst[(size_t)r * dld + c] = (f16)v;
}
__global__ void pad_bias_kernel(const float* __restrict__ b, float* __restrict__ bp,
                                int N, int Np) {
  int i = blockIdx.x * blockDim.x + threadIdx.x;
  if (i < Np) bp[i] = (i < N) ? b[i] : 0.0f;
}
// duplicate px_t into ucat cols [0:512) and [2048:2560)
__global__ void copy_px_kernel(const f16* __restrict__ px, f16* __restrict__ ucat,
                               long long total) {
  long long i = (long long)blockIdx.x * blockDim.x + threadIdx.x;
  if (i >= total) return;
  int r = (int)(i >> 9);
  int c = (int)(i & 511);
  const f16 v = px[i];
  ucat[(size_t)r * UC_LD + c] = v;
  ucat[(size_t)r * UC_LD + 2048 + c] = v;
}
// PyTorch GRUCell gates; h16 written with stride h16ld (into ucat or compact)
__global__ void gru_combine_kernel(const float* __restrict__ gi, const float* __restrict__ gh,
                                   const float* __restrict__ hin,
                                   float* __restrict__ hout, f16* __restrict__ h16,
                                   int h16ld, int Hn, long long total) {
  long long i = (long long)blockIdx.x * blockDim.x + threadIdx.x;
  if (i >= total) return;
  long long b = i / Hn;
  int j = (int)(i % Hn);
  const float* gib = gi + b * 3 * Hn;
  const float* ghb = gh + b * 3 * Hn;
  float r = sigmoidf(gib[j] + ghb[j]);
  float z = sigmoidf(gib[Hn + j] + ghb[Hn + j]);
  float n = tanhf(gib[2 * Hn + j] + r * ghb[2 * Hn + j]);
  float h = (1.0f - z) * n + z * hin[i];
  hout[i] = h;
  h16[b * h16ld + j] = (f16)h;
}
// packed heads P[b,128] = [mean(64) | std_pre(64)]
__global__ void f_sample_kernel(const float* __restrict__ P, const float* __restrict__ eps,
                                f16* __restrict__ fo, float* kld, int n) {
  int i = blockIdx.x * blockDim.x + threadIdx.x;
  float v = 0.0f;
  if (i < n) {
    int b = i >> 6, j = i & 63;
    float m = P[b * 128 + j];
    float s = softplusf(P[b * 128 + 64 + j]);
    fo[i] = (f16)(eps[i] * s + m);
    v = 0.5f * (m * m + s * s - 2.0f * logf(s) - 1.0f);
  }
  block_reduce_add(v, kld);
}
__global__ void z_sample_kernel(const float* __restrict__ Pp, const float* __restrict__ Pe,
                                const float* __restrict__ eps, f16* __restrict__ zo,
                                float* kld, int n) {
  int i = blockIdx.x * blockDim.x + threadIdx.x;
  float v = 0.0f;
  if (i < n) {
    int b = i >> 6, j = i & 63;
    float m2 = Pp[b * 128 + j], s2 = softplusf(Pp[b * 128 + 64 + j]);
    float m1 = Pe[b * 128 + j], s1 = softplusf(Pe[b * 128 + 64 + j]);
    zo[i] = (f16)(eps[i] * s1 + m1);
    float r = s1 / s2;
    float d = (m1 - m2) / s2;
    v = 0.5f * (d * d + r * r - 2.0f * logf(r) - 1.0f);
  }
  block_reduce_add(v, kld);
}
// Bernoulli NLL from pre-sigmoid logits
__global__ void nll_kernel(const float* __restrict__ x, const float* __restrict__ spre,
                           float* nll, int Xc, int lds, int n) {
  int i = blockIdx.x * blockDim.x + threadIdx.x;
  float v = 0.0f;
  if (i < n) {
    int b = i / Xc, c = i % Xc;
    float s = spre[(size_t)b * lds + c];
    float xv = x[i];
    v = xv * softplusf(-s) + (1.0f - xv) * softplusf(s);
  }
  block_reduce_add(v, nll);
}

// ------------------------------------------------------------------
// Host orchestration
// ------------------------------------------------------------------
extern "C" void kernel_launch(void* const* d_in, const int* in_sizes, int n_in,
                              void* d_out, int out_size, void* d_ws, size_t ws_size,
                              hipStream_t stream) {
  (void)in_sizes; (void)n_in; (void)out_size; (void)ws_size;
  float* out = (float*)d_out;
  auto in = [&](int i) { return (const float*)d_in[i]; };

  char* base = (char*)d_ws;
  size_t off = 0;
  auto carve = [&](size_t bytes) -> void* {
    off = (off + 255) & ~(size_t)255;
    void* p = base + off;
    off += bytes;
    return p;
  };
  auto cF16 = [&](size_t n) { return (f16*)carve(n * sizeof(f16)); };
  auto cF32 = [&](size_t n) { return (float*)carve(n * sizeof(float)); };

  f16* xh  = cF16((size_t)TB_DIM * XP_DIM);
  f16* t1h = cF16((size_t)TB_DIM * H_DIM);
  f16* pxh = cF16((size_t)TB_DIM * H_DIM);
  // weights f16, [N, Kp] layout (no transpose)
  f16* WpxW1   = cF16((size_t)H_DIM * XP_DIM);
  f16* WpxW2   = cF16((size_t)H_DIM * H_DIM);
  f16* WpzW    = cF16((size_t)H_DIM * Z_DIM);
  f16* WpfW    = cF16((size_t)H_DIM * F_DIM);
  f16* WrWih   = cF16((size_t)H3_DIM * H3_DIM);
  f16* WrWhh   = cF16((size_t)H3_DIM * H_DIM);
  f16* WzpW    = cF16((size_t)H_DIM * H_DIM);
  f16* Wzpms   = cF16((size_t)128 * H_DIM);
  f16* WfeWih  = cF16((size_t)H3_DIM * H_DIM);
  f16* WfeWhh  = cF16((size_t)H3_DIM * H_DIM);
  f16* Wfemfes = cF16((size_t)128 * H_DIM);
  f16* Wze1    = cF16((size_t)H_DIM * 1024);  // K-order: [h | px]
  f16* WzeW2   = cF16((size_t)H_DIM * H_DIM);
  f16* Wzemzes = cF16((size_t)128 * H_DIM);
  f16* WdW1    = cF16((size_t)H_DIM * H3_DIM); // K-order (phi_z,phi_f,h) = original
  f16* WdW2    = cF16((size_t)H_DIM * H_DIM);
  f16* WdmW    = cF16((size_t)XO_DIM * H_DIM);
  float* dmbp    = cF32(XO_DIM);
  float* bzpms   = cF32(128);
  float* bzemzes = cF32(128);
  float* bfemfes = cF32(128);
  // state / activations
  float* hff32 = cF32((size_t)B_DIM * H_DIM);
  f16*   hff16 = cF16((size_t)B_DIM * H_DIM);
  float* hm32  = cF32((size_t)B_DIM * H_DIM);
  f16*   ucat  = cF16((size_t)B_DIM * UC_LD);
  float* GI    = cF32((size_t)B_DIM * H3_DIM);
  float* GH    = cF32((size_t)B_DIM * H3_DIM);
  float* fpack = cF32((size_t)B_DIM * 128);
  f16*   ff16  = cF16((size_t)B_DIM * F_DIM);
  f16*   zp16  = cF16((size_t)B_DIM * H_DIM);
  float* zppack = cF32((size_t)B_DIM * 128);
  f16*   ze116  = cF16((size_t)B_DIM * H_DIM);
  f16*   ze16   = cF16((size_t)B_DIM * H_DIM);
  float* zepack = cF32((size_t)B_DIM * 128);
  f16*   z16    = cF16((size_t)B_DIM * Z_DIM);
  f16*   dec116 = cF16((size_t)B_DIM * H_DIM);
  f16*   dec16  = cF16((size_t)B_DIM * H_DIM);
  float* dmp    = cF32((size_t)B_DIM * XO_DIM);

  auto gemm = [&](const f16* A, int lda, const f16* W, int ldw, const float* bias,
                  float* C32, f16* C16, int ldc, int M, int N, int K, int act) {
    dim3 g(N / 64, M / 128);
    wmma_gemm_kernel<<<g, 256, 0, stream>>>(A, lda, W, ldw, bias, C32, C16, ldc, K, act);
  };
  auto zf32 = [&](float* p, long long n) {
    zero_f32_kernel<<<(unsigned)((n + 255) / 256), 256, 0, stream>>>(p, n);
  };
  auto zf16 = [&](f16* p, long long n) {
    zero_f16_kernel<<<(unsigned)((n + 255) / 256), 256, 0, stream>>>(p, n);
  };
  auto packw = [&](const float* src, int sld, int rows, int cols,
                   f16* dst, int dld, int dRows, int dCols) {
    long long total = (long long)dRows * dCols;
    pack_w_kernel<<<(unsigned)((total + 255) / 256), 256, 0, stream>>>(
        src, sld, rows, cols, dst, dld, dCols, total);
  };
  auto padb = [&](const float* b, float* bp, int N, int Np) {
    pad_bias_kernel<<<(Np + 255) / 256, 256, 0, stream>>>(b, bp, N, Np);
  };

  const long long BH = (long long)B_DIM * H_DIM;

  // ---- init ----
  zf32(out, 3);
  zf32(hff32, BH); zf16(hff16, BH);
  zf32(hm32, BH);  zf16(ucat, (long long)B_DIM * UC_LD);

  // ---- weight prep (f16 [N,Kp], zero-padded) ----
  packw(in(3),  X_DIM, H_DIM, X_DIM, WpxW1, XP_DIM, H_DIM, XP_DIM);
  packw(in(5),  H_DIM, H_DIM, H_DIM, WpxW2, H_DIM, H_DIM, H_DIM);
  packw(in(7),  Z_DIM, H_DIM, Z_DIM, WpzW, Z_DIM, H_DIM, Z_DIM);
  packw(in(9),  F_DIM, H_DIM, F_DIM, WpfW, F_DIM, H_DIM, F_DIM);
  packw(in(11), H3_DIM, H3_DIM, H3_DIM, WrWih, H3_DIM, H3_DIM, H3_DIM);
  packw(in(12), H_DIM, H3_DIM, H_DIM, WrWhh, H_DIM, H3_DIM, H_DIM);
  packw(in(15), H_DIM, H_DIM, H_DIM, WzpW, H_DIM, H_DIM, H_DIM);
  packw(in(17), H_DIM, 64, H_DIM, Wzpms, H_DIM, 64, H_DIM);
  packw(in(19), H_DIM, 64, H_DIM, Wzpms + (size_t)64 * H_DIM, H_DIM, 64, H_DIM);
  packw(in(21), H_DIM, H3_DIM, H_DIM, WfeWih, H_DIM, H3_DIM, H_DIM);
  packw(in(22), H_DIM, H3_DIM, H_DIM, WfeWhh, H_DIM, H3_DIM, H_DIM);
  packw(in(25), H_DIM, 64, H_DIM, Wfemfes, H_DIM, 64, H_DIM);
  packw(in(27), H_DIM, 64, H_DIM, Wfemfes + (size_t)64 * H_DIM, H_DIM, 64, H_DIM);
  // ze1 weight, K-reordered to [h | px] to match ucat slice [h | px2]
  packw(in(29) + 512, 1024, H_DIM, 512, Wze1, 1024, H_DIM, 512);
  packw(in(29), 1024, H_DIM, 512, Wze1 + 512, 1024, H_DIM, 512);
  packw(in(31), H_DIM, H_DIM, H_DIM, WzeW2, H_DIM, H_DIM, H_DIM);
  packw(in(33), H_DIM, 64, H_DIM, Wzemzes, H_DIM, 64, H_DIM);
  packw(in(35), H_DIM, 64, H_DIM, Wzemzes + (size_t)64 * H_DIM, H_DIM, 64, H_DIM);
  packw(in(37), H3_DIM, H_DIM, H3_DIM, WdW1, H3_DIM, H_DIM, H3_DIM);
  packw(in(39), H_DIM, H_DIM, H_DIM, WdW2, H_DIM, H_DIM, H_DIM);
  packw(in(41), H_DIM, X_DIM, H_DIM, WdmW, H_DIM, XO_DIM, H_DIM);
  padb(in(42), dmbp, X_DIM, XO_DIM);
  padb(in(18), bzpms, 64, 64);      padb(in(20), bzpms + 64, 64, 64);
  padb(in(34), bzemzes, 64, 64);    padb(in(36), bzemzes + 64, 64, 64);
  padb(in(26), bfemfes, 64, 64);    padb(in(28), bfemfes + 64, 64, 64);
  // x f32 [TB,784] -> f16 [TB,800]
  packw(in(0), X_DIM, TB_DIM, X_DIM, xh, XP_DIM, TB_DIM, XP_DIM);

  // ---- stage 1: phi_x ----
  gemm(xh, XP_DIM, WpxW1, XP_DIM, in(4), nullptr, t1h, H_DIM, TB_DIM, H_DIM, XP_DIM, 1);
  gemm(t1h, H_DIM, WpxW2, H_DIM, in(6), nullptr, pxh, H_DIM, TB_DIM, H_DIM, H_DIM, 1);

  // ---- stage 2: f-encoder GRU ----
  for (int t = 0; t < T_DIM; ++t) {
    const f16* px_t = pxh + (size_t)t * B_DIM * H_DIM;
    gemm(px_t, H_DIM, WfeWih, H_DIM, in(23), GI, nullptr, H3_DIM, B_DIM, H3_DIM, H_DIM, 0);
    gemm(hff16, H_DIM, WfeWhh, H_DIM, in(24), GH, nullptr, H3_DIM, B_DIM, H3_DIM, H_DIM, 0);
    gru_combine_kernel<<<(unsigned)((BH + 255) / 256), 256, 0, stream>>>(
        GI, GH, hff32, hff32, hff16, H_DIM, H_DIM, BH);
  }
  gemm(hff16, H_DIM, Wfemfes, H_DIM, bfemfes, fpack, nullptr, 128, B_DIM, 128, H_DIM, 0);
  f_sample_kernel<<<(B_DIM * F_DIM + 255) / 256, 256, 0, stream>>>(
      fpack, in(1), ff16, out + 0, B_DIM * F_DIM);
  // phi_f -> ucat cols [1024:1536), fixed for all t
  gemm(ff16, F_DIM, WpfW, F_DIM, in(10), nullptr, ucat + 1024, UC_LD, B_DIM, H_DIM, F_DIM, 1);

  // ---- stage 3: main VRNN loop ----
  for (int t = 0; t < T_DIM; ++t) {
    const f16* px_t = pxh + (size_t)t * B_DIM * H_DIM;
    const float* epsz_t = in(2) + (size_t)t * B_DIM * Z_DIM;
    const float* x_t = in(0) + (size_t)t * B_DIM * X_DIM;

    copy_px_kernel<<<(unsigned)((BH + 255) / 256), 256, 0, stream>>>(px_t, ucat, BH);
    // prior head (packed mean|std)
    gemm(ucat + 1536, UC_LD, WzpW, H_DIM, in(16), nullptr, zp16, H_DIM, B_DIM, H_DIM, H_DIM, 1);
    gemm(zp16, H_DIM, Wzpms, H_DIM, bzpms, zppack, nullptr, 128, B_DIM, 128, H_DIM, 0);
    // encoder: ze1 = relu([h|px] @ Wze1^T + b) -- single fused-concat GEMM
    gemm(ucat + 1536, UC_LD, Wze1, 1024, in(30), nullptr, ze116, H_DIM, B_DIM, H_DIM, 1024, 1);
    gemm(ze116, H_DIM, WzeW2, H_DIM, in(32), nullptr, ze16, H_DIM, B_DIM, H_DIM, H_DIM, 1);
    gemm(ze16, H_DIM, Wzemzes, H_DIM, bzemzes, zepack, nullptr, 128, B_DIM, 128, H_DIM, 0);
    z_sample_kernel<<<(B_DIM * Z_DIM + 255) / 256, 256, 0, stream>>>(
        zppack, zepack, epsz_t, z16, out + 1, B_DIM * Z_DIM);
    // phi_z -> ucat cols [512:1024)
    gemm(z16, Z_DIM, WpzW, Z_DIM, in(8), nullptr, ucat + 512, UC_LD, B_DIM, H_DIM, Z_DIM, 1);
    // decoder: dec1 = relu([phi_z|phi_f|h] @ dW1^T + b) -- single GEMM K=1536
    gemm(ucat + 512, UC_LD, WdW1, H3_DIM, in(38), nullptr, dec116, H_DIM, B_DIM, H_DIM, H3_DIM, 1);
    gemm(dec116, H_DIM, WdW2, H_DIM, in(40), nullptr, dec16, H_DIM, B_DIM, H_DIM, H_DIM, 1);
    gemm(dec16, H_DIM, WdmW, H_DIM, dmbp, dmp, nullptr, XO_DIM, B_DIM, XO_DIM, H_DIM, 0);
    nll_kernel<<<(B_DIM * X_DIM + 255) / 256, 256, 0, stream>>>(
        x_t, dmp, out + 2, X_DIM, XO_DIM, B_DIM * X_DIM);
    // recurrence GRU: gi over [px|phi_z|phi_f] (K=1536, single GEMM), gh over h
    gemm(ucat, UC_LD, WrWih, H3_DIM, in(13), GI, nullptr, H3_DIM, B_DIM, H3_DIM, H3_DIM, 0);
    gemm(ucat + 1536, UC_LD, WrWhh, H_DIM, in(14), GH, nullptr, H3_DIM, B_DIM, H3_DIM, H_DIM, 0);
    gru_combine_kernel<<<(unsigned)((BH + 255) / 256), 256, 0, stream>>>(
        GI, GH, hm32, hm32, ucat + 1536, UC_LD, H_DIM, BH);
  }
}
